// DynamicMoE_16776142258501
// MI455X (gfx1250) — matrix operand order repaired and checked by reading, not compiled, not run on hardware
//
#include <hip/hip_runtime.h>
#include <hip/hip_bf16.h>

// Problem constants (B,S,D,E,K) = (4,2048,1024,8,2), H=4096
#define NT   8192      // B*S tokens
#define DD   1024
#define HH   4096
#define EE   8
#define MAXT NT
#define BM   128
#define BN   128
#define BK   32
#define LDSK 40        // padded LDS row stride (ushorts): 80B, 16B-aligned, bank-spread

typedef __attribute__((ext_vector_type(16))) __bf16 v16bf;
typedef __attribute__((ext_vector_type(8)))  float  v8f;
typedef __attribute__((ext_vector_type(4)))  int    v4i;

#if defined(__has_builtin)
# if __has_builtin(__builtin_amdgcn_global_load_async_to_lds_b128)
#  define USE_ASYNC_LDS 1
# endif
# if __has_builtin(__builtin_amdgcn_s_wait_asynccnt)
#  define HAVE_WAIT_ASYNC 1
# endif
#endif

__device__ __forceinline__ unsigned short f2bf(float f) {
    return __builtin_bit_cast(unsigned short, (__bf16)f);
}

// 16-byte tile chunk mover: async DMA to LDS when available, else load+ds_store.
__device__ __forceinline__ void copy16_to_lds(const unsigned short* g, unsigned short* l) {
#ifdef USE_ASYNC_LDS
    __builtin_amdgcn_global_load_async_to_lds_b128(
        (__attribute__((address_space(1))) v4i*)(unsigned long long)g,  // global AS is numerically flat
        (__attribute__((address_space(3))) v4i*)l,                      // addrspacecast generic->LDS
        0, 0);
#else
    *(uint4*)l = *(const uint4*)g;
#endif
}

__device__ __forceinline__ void wait_tiles() {
#ifdef USE_ASYNC_LDS
# ifdef HAVE_WAIT_ASYNC
    __builtin_amdgcn_s_wait_asynccnt(0);
# else
    asm volatile("s_wait_asynccnt 0x0" ::: "memory");
# endif
#endif
}

// ---------------------------------------------------------------- init
__global__ void moe_init(int* cnt) {
    if (threadIdx.x < EE) cnt[threadIdx.x] = 0;
}

// ---------------------------------------------------------------- fp32 [E][K][N] -> bf16 [E][N][K] (convert + transpose)
// Transposing here makes GEMM B-tiles contiguous-K: LDS fill is pure 16B moves.
__global__ __launch_bounds__(256)
void moe_cvt_tr(const float* __restrict__ src, unsigned short* __restrict__ dst,
                int Kd, int Nd) {
    __shared__ unsigned short tile[32][33];
    const int e  = blockIdx.z;
    const int k0 = blockIdx.x * 32, n0 = blockIdx.y * 32;
    const int tx = threadIdx.x & 31, ty = threadIdx.x >> 5;   // 32 x 8
    const float* s = src + (size_t)e * Kd * Nd;
#pragma unroll
    for (int i = 0; i < 4; ++i) {
        int k = ty * 4 + i;
        tile[tx][k] = f2bf(s[(size_t)(k0 + k) * Nd + n0 + tx]);   // coalesced in n
    }
    __syncthreads();
    unsigned short* d = dst + (size_t)e * Kd * Nd;
#pragma unroll
    for (int i = 0; i < 4; ++i) {
        int n = ty * 4 + i;
        d[(size_t)(n0 + n) * Kd + k0 + tx] = tile[n][tx];          // coalesced in k
    }
}

// ---------------------------------------------------------------- gating: one wave per token
__global__ __launch_bounds__(256)
void moe_gate(const float* __restrict__ x, const float* __restrict__ gw,
              const float* __restrict__ gb, unsigned short* __restrict__ xb,
              int* __restrict__ cnt, int* __restrict__ perm) {
    int wid = threadIdx.x >> 5, lane = threadIdx.x & 31;
    int token = blockIdx.x * 8 + wid;
    const float* xrow = x + (size_t)token * DD;

    float acc[EE];
    float xreg[DD / 32];
#pragma unroll
    for (int e = 0; e < EE; ++e) acc[e] = 0.f;
#pragma unroll
    for (int i = 0; i < DD / 32; ++i) {
        int d = i * 32 + lane;
        float xv = xrow[d];
        xreg[i] = xv;
        const float4* g = (const float4*)(gw + (size_t)d * EE);
        float4 g0 = g[0], g1 = g[1];
        acc[0] += xv * g0.x; acc[1] += xv * g0.y;
        acc[2] += xv * g0.z; acc[3] += xv * g0.w;
        acc[4] += xv * g1.x; acc[5] += xv * g1.y;
        acc[6] += xv * g1.z; acc[7] += xv * g1.w;
    }
#pragma unroll
    for (int off = 16; off >= 1; off >>= 1)
#pragma unroll
        for (int e = 0; e < EE; ++e) acc[e] += __shfl_xor(acc[e], off, 32);

    float m = -1e30f;
#pragma unroll
    for (int e = 0; e < EE; ++e) { acc[e] += gb[e]; m = fmaxf(m, acc[e]); }
    float s = 0.f;
#pragma unroll
    for (int e = 0; e < EE; ++e) { acc[e] = __expf(acc[e] - m); s += acc[e]; }
    float inv = 1.f / s;

    int   i1 = 0; float p1 = acc[0];
#pragma unroll
    for (int e = 1; e < EE; ++e) if (acc[e] > p1) { p1 = acc[e]; i1 = e; }
    int   i2 = -1; float p2 = -1.f;
#pragma unroll
    for (int e = 0; e < EE; ++e) if (e != i1 && acc[e] > p2) { p2 = acc[e]; i2 = e; }
    // "where" overwrite semantics: largest-index selected expert wins
    int   wexp  = (i1 > i2) ? i1 : i2;
    float scale = ((wexp == i1) ? p1 : p2) * inv;

    unsigned short* xo = xb + (size_t)token * DD;
#pragma unroll
    for (int i = 0; i < DD / 32; ++i) xo[i * 32 + lane] = f2bf(xreg[i] * scale);

    if (lane == 0) {
        int pos = atomicAdd(&cnt[wexp], 1);
        perm[wexp * MAXT + pos] = token;
    }
}

// ---------------------------------------------------------------- expert GEMM (both layers)
// A:  [NT][Kdim] bf16 (rows gathered by perm)
// Wt: [E][Ndim][Kdim] bf16 (pre-transposed -> contiguous K)
// FIRST:  hout[tok][Ndim] = bf16(relu(A@W + b1));  !FIRST: yout[tok][Ndim] = A@W + b2
template <bool FIRST>
__global__ __launch_bounds__(256)
void moe_gemm(const unsigned short* __restrict__ A,
              const unsigned short* __restrict__ Wt,
              const float* __restrict__ bias,
              const int* __restrict__ perm, const int* __restrict__ cnt,
              unsigned short* __restrict__ hout, float* __restrict__ yout,
              int Kdim, int Ndim) {
    __shared__ unsigned short At[2][BM * LDSK];   // [m][k], double-buffered
    __shared__ unsigned short Bt[2][BN * LDSK];   // [n][k], double-buffered
    __shared__ int tokrow[BM];

    const int e  = blockIdx.z;
    const int n  = cnt[e];
    const int mt = blockIdx.x;
    if (mt * BM >= n) return;                     // dynamic early-exit on routed count
    const int n0 = blockIdx.y * BN;

    const int t = threadIdx.x, lane = t & 31, wid = t >> 5;
    const int waveM = wid & 3, waveN = wid >> 2;  // 4 x 2 waves -> 32x64 per wave

    if (t < BM) {
        int gm = mt * BM + t;
        int tok = (gm < n) ? perm[e * MAXT + gm] : 0;  // padded rows load token 0 (harmless: stores guarded)
        tokrow[t] = tok;
    }
    __syncthreads();

    const unsigned short* We = Wt + (size_t)e * Kdim * Ndim;

    // per-thread chunk coordinates (512 x 16B chunks per 128x32 tile, 2 per thread)
    const int row0 = t >> 2, kc0 = (t & 3) * 8;
    const int row1 = row0 + 64, kc1 = kc0;
    const int tokA0 = tokrow[row0], tokA1 = tokrow[row1];

    auto issue_tile = [&](int k0, int buf) {
        copy16_to_lds(A + (size_t)tokA0 * Kdim + k0 + kc0, &At[buf][row0 * LDSK + kc0]);
        copy16_to_lds(A + (size_t)tokA1 * Kdim + k0 + kc1, &At[buf][row1 * LDSK + kc1]);
        const unsigned short* b0 = We + (size_t)(n0 + row0) * Kdim + k0 + kc0;
        const unsigned short* b1 = We + (size_t)(n0 + row1) * Kdim + k0 + kc1;
        copy16_to_lds(b0, &Bt[buf][row0 * LDSK + kc0]);
        copy16_to_lds(b1, &Bt[buf][row1 * LDSK + kc1]);
        if (k0 + BK < Kdim) __builtin_prefetch(b0 + BK, 0, 1);   // global_prefetch_b8, next K-tile
    };

    v8f acc[2][4];
    v8f zero = {};
#pragma unroll
    for (int a = 0; a < 2; ++a)
#pragma unroll
        for (int b = 0; b < 4; ++b) acc[a][b] = zero;

    const int nk = Kdim / BK;
    issue_tile(0, 0);

    for (int kt = 0; kt < nk; ++kt) {
        wait_tiles();            // this wave's outstanding tile landed in LDS
        __syncthreads();         // all waves' pieces visible; prev compute done
        if (kt + 1 < nk) issue_tile((kt + 1) * BK, (kt + 1) & 1);   // overlap copy with WMMA below

        const unsigned short* Ac = At[kt & 1];
        const unsigned short* Bc = Bt[kt & 1];

        union Frag { uint4 u[2]; v16bf v; };
        Frag fa[2];
        const int kbA = (lane >> 4) * 8;          // A: elems 0-7 -> K=kbA.., 8-15 -> K=kbA+16..
#pragma unroll
        for (int a = 0; a < 2; ++a) {
            int row = waveM * 32 + a * 16 + (lane & 15);
            fa[a].u[0] = *(const uint4*)(&Ac[row * LDSK + kbA]);
            fa[a].u[1] = *(const uint4*)(&Ac[row * LDSK + kbA + 16]);
        }
        const int kbB = (lane >> 4) * 8;          // B: elems 0-7 -> K=kbB.., 8-15 -> K=kbB+16..
#pragma unroll
        for (int b = 0; b < 4; ++b) {
            Frag fb;
            int col = waveN * 64 + b * 16 + (lane & 15);
            fb.u[0] = *(const uint4*)(&Bc[col * LDSK + kbB]);
            fb.u[1] = *(const uint4*)(&Bc[col * LDSK + kbB + 16]);
#pragma unroll
            for (int a = 0; a < 2; ++a)
                acc[a][b] = __builtin_amdgcn_wmma_f32_16x16x32_bf16(
                    false, fa[a].v, false, fb.v, (short)0, acc[a][b], false, false);
        }
    }

    // epilogue: C/D layout -> M = r + 8*(lane>=16), N = lane&15
#pragma unroll
    for (int b = 0; b < 4; ++b) {
        int col = n0 + waveN * 64 + b * 16 + (lane & 15);
        float bv = bias[e * Ndim + col];
#pragma unroll
        for (int a = 0; a < 2; ++a) {
#pragma unroll
            for (int r = 0; r < 8; ++r) {
                int rowin = waveM * 32 + a * 16 + r + ((lane >> 4) << 3);
                int gm = mt * BM + rowin;
                if (gm < n) {
                    int tok = tokrow[rowin];
                    float v = acc[a][b][r] + bv;
                    if (FIRST) {
                        v = fmaxf(v, 0.f);
                        hout[(size_t)tok * Ndim + col] = f2bf(v);
                    } else {
                        yout[(size_t)tok * Ndim + col] = v;
                    }
                }
            }
        }
    }
}

// ---------------------------------------------------------------- host launcher
extern "C" void kernel_launch(void* const* d_in, const int* in_sizes, int n_in,
                              void* d_out, int out_size, void* d_ws, size_t ws_size,
                              hipStream_t stream) {
    const float* x  = (const float*)d_in[0];
    const float* gw = (const float*)d_in[1];
    const float* gb = (const float*)d_in[2];
    const float* w1 = (const float*)d_in[3];
    const float* b1 = (const float*)d_in[4];
    const float* w2 = (const float*)d_in[5];
    const float* b2 = (const float*)d_in[6];
    float* out = (float*)d_out;

    // workspace carve-up (~208.3 MB total)
    char* ws = (char*)d_ws;
    int* cnt  = (int*)ws;                                  // 8 ints (pad 256B)
    int* perm = (int*)(ws + 256);                          // E*MAXT ints = 256 KB
    unsigned short* xb   = (unsigned short*)(ws + 256 + (size_t)EE * MAXT * 4);
    unsigned short* hbuf = xb   + (size_t)NT * DD;         // xb 16 MB
    unsigned short* w1t  = hbuf + (size_t)NT * HH;         // hbuf 64 MB
    unsigned short* w2t  = w1t  + (size_t)EE * DD * HH;    // w1t 64 MB, w2t 64 MB

    moe_init<<<1, 32, 0, stream>>>(cnt);

    // convert + transpose weights: w1 [E][D][H] -> w1t [E][H][D]; w2 [E][H][D] -> w2t [E][D][H]
    dim3 c1(DD / 32, HH / 32, EE);
    moe_cvt_tr<<<c1, 256, 0, stream>>>(w1, w1t, DD, HH);
    dim3 c2(HH / 32, DD / 32, EE);
    moe_cvt_tr<<<c2, 256, 0, stream>>>(w2, w2t, HH, DD);

    moe_gate<<<NT / 8, 256, 0, stream>>>(x, gw, gb, xb, cnt, perm);

    // worst-case grid (one expert owns all tokens); blocks early-exit on cnt[e]
    dim3 g1(MAXT / BM, HH / BN, EE);
    moe_gemm<true><<<g1, 256, 0, stream>>>(xb, w1t, b1, perm, cnt, hbuf, (float*)nullptr, DD, HH);

    dim3 g2(MAXT / BM, DD / BN, EE);
    moe_gemm<false><<<g2, 256, 0, stream>>>(hbuf, w2t, b2, perm, cnt, (unsigned short*)nullptr, out, HH, DD);
}
